// GATGraphClassifier_39848706573596
// MI455X (gfx1250) — matrix operand order repaired and checked by reading, not compiled, not run on hardware
//
#include <hip/hip_runtime.h>
#include <hip/hip_bf16.h>

// ---------------------------------------------------------------------------
// GAT graph classifier for MI455X (gfx1250, wave32, WMMA).
//
// Roofline: the two GEMMs (6.6 + 3.3 GFLOP) run on v_wmma_f32_16x16x32_f16;
// the edge gather/scatter (~3.2 GB of irregular traffic per call) is the HBM
// roofline floor and is implemented as streaming passes with hardware float
// atomics (global_atomic_add_f32 via unsafeAtomicAdd, global_atomic_max_u32
// with an order-preserving u32 encoding for segment-max).
//
// GEMM: one wave computes a 16x64 strip of C (4 N-tiles, 4 f32 accumulators).
// The K-loop is software-pipelined: next step's A + 4 B tiles are prefetched
// into a second register set while the current step's 4 WMMAs issue, so the
// per-WMMA s_wait_loadcnt 0x0 bubble is replaced by loads a full iteration
// ahead of their consuming WMMA.
// ---------------------------------------------------------------------------

typedef __attribute__((ext_vector_type(16))) _Float16 v16h;
typedef __attribute__((ext_vector_type(8)))  _Float16 v8h;
typedef __attribute__((ext_vector_type(4)))  _Float16 v4h;
typedef __attribute__((ext_vector_type(8)))  float    v8f;
typedef __attribute__((ext_vector_type(4)))  float    v4f;

#define NNODES   50000
#define NEDGES   800000
#define INCH     128
#define HID      64
#define HEADS    8
#define NGRAPHS  512
#define OUTCH    10
#define NEGSLOPE 0.2f

// ---------------- helpers ----------------

__device__ __forceinline__ unsigned f2ord(float f) {
    unsigned u = __float_as_uint(f);
    return (u & 0x80000000u) ? ~u : (u | 0x80000000u);
}

// ---------------- elementwise / pack kernels ----------------

// 4-wide f32 -> f16 convert (n multiple of 4)
__global__ void cvt_f16(const float* __restrict__ in, _Float16* __restrict__ out, int n4) {
    int t = blockIdx.x * blockDim.x + threadIdx.x;
    if (t >= n4) return;
    v4f v = *(const v4f*)(in + (size_t)t * 4);
    v4h o;
    o[0] = (_Float16)v[0]; o[1] = (_Float16)v[1];
    o[2] = (_Float16)v[2]; o[3] = (_Float16)v[3];
    *(v4h*)(out + (size_t)t * 4) = o;
}

// in: [Kdim, Ndim] row-major -> out: [Ndim, Kdim] row-major, f16
__global__ void transpose_cvt(const float* __restrict__ in, _Float16* __restrict__ out,
                              int Kdim, int Ndim) {
    int t = blockIdx.x * blockDim.x + threadIdx.x;
    if (t >= Kdim * Ndim) return;
    int n = t / Kdim, k = t % Kdim;
    out[t] = (_Float16)in[(size_t)k * Ndim + n];
}

// ---------------- WMMA GEMM: C[M,N] = A[M,K](f16) * Bt[N,K]^T(f16) ----------------
// One wave per 16x(16*NTW) strip of C; K multiple of 32; Ntiles multiple of NTW.
// Software-pipelined K-loop (double-buffered A/B register sets).

template <int K, int NTW>
__global__ __launch_bounds__(256) void wmma_gemm(const _Float16* __restrict__ A,
                                                 const _Float16* __restrict__ Bt,
                                                 float* __restrict__ C,
                                                 int Mtiles, int Ntiles) {
    int wave = (blockIdx.x * blockDim.x + threadIdx.x) >> 5;
    int lane = threadIdx.x & 31;
    int ngroups = Ntiles / NTW;
    if (wave >= Mtiles * ngroups) return;
    int mt  = wave / ngroups;
    int nt0 = (wave % ngroups) * NTW;
    int hl  = lane >> 4;      // half-wave selector
    int lr  = lane & 15;
    int N   = Ntiles * 16;

    const _Float16* arow = A  + (size_t)(mt * 16 + lr) * K;          // A row M=lr
    const _Float16* brow = Bt + (size_t)(nt0 * 16 + lr) * K;         // B col N=nt0*16+lr

    // A 16-bit layout: lanes0-15 K=[k0..k0+7]&[k0+16..k0+23]; lanes16-31 +8
    auto loadA = [&](int k0) -> v16h {
        v16h a;
        ((v8h*)&a)[0] = *(const v8h*)(arow + k0 + hl * 8);
        ((v8h*)&a)[1] = *(const v8h*)(arow + k0 + 16 + hl * 8);
        return a;
    };
    // B 16-bit layout: same K order as A (lanes0-15 K=[k0..k0+7]&[k0+16..k0+23]; lanes16-31 +8)
    auto loadB = [&](int j, int k0) -> v16h {
        v16h b;
        ((v8h*)&b)[0] = *(const v8h*)(brow + (size_t)j * 16 * K + k0 + hl * 8);
        ((v8h*)&b)[1] = *(const v8h*)(brow + (size_t)j * 16 * K + k0 + 16 + hl * 8);
        return b;
    };

    v8f acc[NTW];
#pragma unroll
    for (int j = 0; j < NTW; ++j) acc[j] = (v8f){};

    // prologue
    v16h a_cur = loadA(0);
    v16h b_cur[NTW];
#pragma unroll
    for (int j = 0; j < NTW; ++j) b_cur[j] = loadB(j, 0);

#pragma unroll
    for (int k0 = 0; k0 < K; k0 += 32) {
        v16h a_nxt;
        v16h b_nxt[NTW];
        if (k0 + 32 < K) {
            a_nxt = loadA(k0 + 32);
#pragma unroll
            for (int j = 0; j < NTW; ++j) b_nxt[j] = loadB(j, k0 + 32);
        }
#pragma unroll
        for (int j = 0; j < NTW; ++j)
            acc[j] = __builtin_amdgcn_wmma_f32_16x16x32_f16(false, a_cur, false, b_cur[j],
                                                            (short)0, acc[j], false, false);
        a_cur = a_nxt;
#pragma unroll
        for (int j = 0; j < NTW; ++j) b_cur[j] = b_nxt[j];
    }

    // C layout: VGPR r -> row m0 + r + 8*half, col n0 + lr
#pragma unroll
    for (int j = 0; j < NTW; ++j) {
        float* cp = C + (size_t)(mt * 16 + hl * 8) * N + (nt0 + j) * 16 + lr;
#pragma unroll
        for (int r = 0; r < 8; ++r) cp[(size_t)r * N] = acc[j][r];
    }
}

// ---------------- attention scores: a_s/a_d [N,H] ----------------

__global__ void attn_scores(const float* __restrict__ xh,   // [N, H*64]
                            const float* __restrict__ asv,  // [H,64]
                            const float* __restrict__ adv,  // [H,64]
                            float* __restrict__ a_s, float* __restrict__ a_d,
                            int nNodes, int H) {
    int t = blockIdx.x * blockDim.x + threadIdx.x;
    if (t >= nNodes * H) return;
    int h = t % H;
    const v4f* xp = (const v4f*)(xh + (size_t)t * 64);  // t = node*H+h -> node*H*64 + h*64
    const v4f* sp = (const v4f*)(asv + h * 64);
    const v4f* dp = (const v4f*)(adv + h * 64);
    float s = 0.f, d = 0.f;
#pragma unroll
    for (int i = 0; i < 16; ++i) {
        v4f v = xp[i], sv = sp[i], dv = dp[i];
        s += v[0] * sv[0] + v[1] * sv[1] + v[2] * sv[2] + v[3] * sv[3];
        d += v[0] * dv[0] + v[1] * dv[1] + v[2] * dv[2] + v[3] * dv[3];
    }
    a_s[t] = s; a_d[t] = d;
}

// ---------------- edge pass 1: segment max (order-preserving u32) ----------------

template <int H>
__global__ void edge_max(const int* __restrict__ src, const int* __restrict__ dst,
                         const float* __restrict__ a_s, const float* __restrict__ a_d,
                         unsigned* __restrict__ emaxk, int nE) {
    int t = blockIdx.x * blockDim.x + threadIdx.x;
    if (t >= nE * H) return;
    int e = t / H, h = t % H;     // H is a power of two -> shift/mask
    int s = src[e], d = dst[e];
    float v = a_s[(size_t)s * H + h] + a_d[(size_t)d * H + h];
    v = v > 0.f ? v : NEGSLOPE * v;
    atomicMax(&emaxk[(size_t)d * H + h], f2ord(v));
}

__global__ void decode_max(unsigned* __restrict__ k, int n) {
    int t = blockIdx.x * blockDim.x + threadIdx.x;
    if (t >= n) return;
    unsigned u = k[t];
    float f;
    if (u == 0u) f = 0.0f;                                  // no in-edges -> 0 (matches ref)
    else if (u & 0x80000000u) f = __uint_as_float(u & 0x7FFFFFFFu);
    else f = __uint_as_float(~u);
    ((float*)k)[t] = f;
}

// ---------------- edge pass 2: exp + denominator ----------------

template <int H>
__global__ void edge_exp(const int* __restrict__ src, const int* __restrict__ dst,
                         const float* __restrict__ a_s, const float* __restrict__ a_d,
                         const float* __restrict__ emax, float* __restrict__ ee,
                         float* __restrict__ denom, int nE) {
    int t = blockIdx.x * blockDim.x + threadIdx.x;
    if (t >= nE * H) return;
    int e = t / H, h = t % H;
    int s = src[e], d = dst[e];
    float v = a_s[(size_t)s * H + h] + a_d[(size_t)d * H + h];
    v = v > 0.f ? v : NEGSLOPE * v;
    float ex = __expf(v - emax[(size_t)d * H + h]);
    ee[t] = ex;
    unsafeAtomicAdd(&denom[(size_t)d * H + h], ex);
}

// ---------------- edge pass 3: alpha-weighted gather/scatter ----------------
// One wave per edge; CH = H*64 channels; each lane handles CH/32 consecutive
// channels (16 for layer1, 2 for layer2) -> exactly one head per lane.

template <int H, int CH>
__global__ void edge_msg(const int* __restrict__ src, const int* __restrict__ dst,
                         const float* __restrict__ xh, const float* __restrict__ ee,
                         const float* __restrict__ denom, float* __restrict__ out,
                         int nE) {
    int wave = (blockIdx.x * blockDim.x + threadIdx.x) >> 5;
    int lane = threadIdx.x & 31;
    if (wave >= nE) return;
    int s = src[wave], d = dst[wave];
    constexpr int chunk = CH / 32;
    int c0 = lane * chunk;
    int h  = c0 >> 6;
    float alpha = ee[(size_t)wave * H + h] / (denom[(size_t)d * H + h] + 1e-16f);
    const float* xp = xh + (size_t)s * CH + c0;
    float* op = out + (size_t)d * CH + c0;
#pragma unroll
    for (int j = 0; j < chunk; ++j)
        unsafeAtomicAdd(op + j, xp[j] * alpha);
}

// ---------------- bias + relu + f16 convert (layer1 output), 4-wide ----------------

__global__ void bias_relu_cvt(const float* __restrict__ in, const float* __restrict__ b,
                              _Float16* __restrict__ out, int n4, int C) {
    int t = blockIdx.x * blockDim.x + threadIdx.x;
    if (t >= n4) return;
    int base = t * 4;
    v4f v = *(const v4f*)(in + (size_t)base);
    v4f bv = *(const v4f*)(b + (base % C));
    v4h o;
#pragma unroll
    for (int j = 0; j < 4; ++j) {
        float r = v[j] + bv[j];
        o[j] = (_Float16)(r > 0.f ? r : 0.f);
    }
    *(v4h*)(out + (size_t)base) = o;
}

// ---------------- batch pooling with fused bias ----------------

__global__ void pool_sum(const float* __restrict__ h, const float* __restrict__ b2,
                         const int* __restrict__ batch, float* __restrict__ pooled, int n) {
    int t = blockIdx.x * blockDim.x + threadIdx.x;
    if (t >= n) return;
    int node = t >> 6, c = t & 63;
    int g = batch[node];
    unsafeAtomicAdd(&pooled[(size_t)g * 64 + c], h[t] + b2[c]);
}

// ---------------- FC head + log_softmax ----------------

__global__ void fc_head(const float* __restrict__ pooled, const float* __restrict__ fcw,
                        const float* __restrict__ fcb, float* __restrict__ out) {
    int g = blockIdx.x * blockDim.x + threadIdx.x;
    if (g >= NGRAPHS) return;
    float l[OUTCH];
#pragma unroll
    for (int o = 0; o < OUTCH; ++o) l[o] = fcb[o];
    for (int c = 0; c < HID; ++c) {
        float p = pooled[(size_t)g * HID + c];
#pragma unroll
        for (int o = 0; o < OUTCH; ++o) l[o] += p * fcw[c * OUTCH + o];
    }
    float m = l[0];
#pragma unroll
    for (int o = 1; o < OUTCH; ++o) m = fmaxf(m, l[o]);
    float sum = 0.f;
#pragma unroll
    for (int o = 0; o < OUTCH; ++o) sum += __expf(l[o] - m);
    float lse = m + __logf(sum);
#pragma unroll
    for (int o = 0; o < OUTCH; ++o) out[(size_t)g * OUTCH + o] = l[o] - lse;
}

// ---------------------------------------------------------------------------

static inline int nblk(long n, int b) { return (int)((n + b - 1) / b); }

extern "C" void kernel_launch(void* const* d_in, const int* in_sizes, int n_in,
                              void* d_out, int out_size, void* d_ws, size_t ws_size,
                              hipStream_t stream) {
    const float* x        = (const float*)d_in[0];
    const int*   eidx     = (const int*)d_in[1];     // [2, E]
    const int*   batch    = (const int*)d_in[2];
    const float* W1       = (const float*)d_in[3];   // [128, 512]
    const float* att_s1   = (const float*)d_in[4];   // [8, 64]
    const float* att_d1   = (const float*)d_in[5];
    const float* b1       = (const float*)d_in[6];   // [512]
    const float* W2       = (const float*)d_in[7];   // [512, 64]
    const float* att_s2   = (const float*)d_in[8];   // [1, 64]
    const float* att_d2   = (const float*)d_in[9];
    const float* b2       = (const float*)d_in[10];  // [64]
    const float* fcw      = (const float*)d_in[11];  // [64, 10]
    const float* fcb      = (const float*)d_in[12];  // [10]
    float* out = (float*)d_out;

    const int* src = eidx;
    const int* dst = eidx + NEDGES;

    // ---- workspace carve (256B aligned) ----
    size_t off = 0;
    char* base = (char*)d_ws;
    auto alloc = [&](size_t bytes) -> char* {
        char* p = base + off;
        off += (bytes + 255) & ~(size_t)255;
        return p;
    };
    _Float16* x16   = (_Float16*)alloc((size_t)NNODES * INCH * 2);
    _Float16* W1t   = (_Float16*)alloc((size_t)512 * 128 * 2);      // [N=512, K=128]
    _Float16* W2t   = (_Float16*)alloc((size_t)64 * 512 * 2);       // [N=64,  K=512]
    float*    xh1   = (float*)alloc((size_t)NNODES * 512 * 4);
    float*    as1   = (float*)alloc((size_t)NNODES * HEADS * 4);
    float*    ad1   = (float*)alloc((size_t)NNODES * HEADS * 4);
    unsigned* emx1  = (unsigned*)alloc((size_t)NNODES * HEADS * 4);
    float*    den1  = (float*)alloc((size_t)NNODES * HEADS * 4);
    float*    ee1   = (float*)alloc((size_t)NEDGES * HEADS * 4);
    float*    out1  = (float*)alloc((size_t)NNODES * 512 * 4);
    _Float16* h1h   = (_Float16*)alloc((size_t)NNODES * 512 * 2);
    float*    xh2   = (float*)alloc((size_t)NNODES * 64 * 4);
    float*    as2   = (float*)alloc((size_t)NNODES * 4);
    float*    ad2   = (float*)alloc((size_t)NNODES * 4);
    unsigned* emx2  = (unsigned*)alloc((size_t)NNODES * 4);
    float*    den2  = (float*)alloc((size_t)NNODES * 4);
    float*    ee2   = (float*)alloc((size_t)NEDGES * 4);
    float*    out2  = (float*)alloc((size_t)NNODES * 64 * 4);
    float*    pool  = (float*)alloc((size_t)NGRAPHS * 64 * 4);
    (void)ws_size; (void)n_in; (void)in_sizes; (void)out_size;

    const int B = 256;

    // zero accumulators (async memset is graph-capturable)
    hipMemsetAsync(emx1, 0, (size_t)NNODES * HEADS * 4, stream);
    hipMemsetAsync(den1, 0, (size_t)NNODES * HEADS * 4, stream);
    hipMemsetAsync(out1, 0, (size_t)NNODES * 512 * 4, stream);
    hipMemsetAsync(emx2, 0, (size_t)NNODES * 4, stream);
    hipMemsetAsync(den2, 0, (size_t)NNODES * 4, stream);
    hipMemsetAsync(out2, 0, (size_t)NNODES * 64 * 4, stream);
    hipMemsetAsync(pool, 0, (size_t)NGRAPHS * 64 * 4, stream);

    // ---- pack inputs to f16 ----
    cvt_f16<<<nblk((long)NNODES * INCH / 4, B), B, 0, stream>>>(x, x16, NNODES * INCH / 4);
    transpose_cvt<<<nblk(512 * 128, B), B, 0, stream>>>(W1, W1t, 128, 512);
    transpose_cvt<<<nblk(64 * 512, B), B, 0, stream>>>(W2, W2t, 512, 64);

    // ---- layer 1: GEMM (50000x128 @ 128x512), WMMA f16->f32 ----
    {
        int Mt = NNODES / 16, Nt = 512 / 16;                // 3125 x 32 tiles
        long waves = (long)Mt * (Nt / 4);                   // 16x64 strip per wave
        wmma_gemm<128, 4><<<nblk(waves * 32, B), B, 0, stream>>>(x16, W1t, xh1, Mt, Nt);
    }
    attn_scores<<<nblk((long)NNODES * HEADS, B), B, 0, stream>>>(xh1, att_s1, att_d1,
                                                                 as1, ad1, NNODES, HEADS);
    edge_max<HEADS><<<nblk((long)NEDGES * HEADS, B), B, 0, stream>>>(src, dst, as1, ad1,
                                                                     emx1, NEDGES);
    decode_max<<<nblk((long)NNODES * HEADS, B), B, 0, stream>>>(emx1, NNODES * HEADS);
    edge_exp<HEADS><<<nblk((long)NEDGES * HEADS, B), B, 0, stream>>>(src, dst, as1, ad1,
                                                                     (const float*)emx1,
                                                                     ee1, den1, NEDGES);
    edge_msg<HEADS, 512><<<nblk((long)NEDGES * 32, B), B, 0, stream>>>(src, dst, xh1, ee1,
                                                                       den1, out1, NEDGES);
    bias_relu_cvt<<<nblk((long)NNODES * 512 / 4, B), B, 0, stream>>>(out1, b1, h1h,
                                                                     NNODES * 512 / 4, 512);

    // ---- layer 2: GEMM (50000x512 @ 512x64), WMMA f16->f32 ----
    {
        int Mt = NNODES / 16, Nt = 64 / 16;                 // 3125 x 4 tiles
        long waves = (long)Mt * (Nt / 4);                   // one 16x64 strip per wave
        wmma_gemm<512, 4><<<nblk(waves * 32, B), B, 0, stream>>>(h1h, W2t, xh2, Mt, Nt);
    }
    attn_scores<<<nblk((long)NNODES, B), B, 0, stream>>>(xh2, att_s2, att_d2,
                                                         as2, ad2, NNODES, 1);
    edge_max<1><<<nblk((long)NEDGES, B), B, 0, stream>>>(src, dst, as2, ad2, emx2, NEDGES);
    decode_max<<<nblk((long)NNODES, B), B, 0, stream>>>(emx2, NNODES);
    edge_exp<1><<<nblk((long)NEDGES, B), B, 0, stream>>>(src, dst, as2, ad2,
                                                         (const float*)emx2, ee2, den2,
                                                         NEDGES);
    edge_msg<1, 64><<<nblk((long)NEDGES * 32, B), B, 0, stream>>>(src, dst, xh2, ee2, den2,
                                                                  out2, NEDGES);

    // ---- pool + head ----
    pool_sum<<<nblk((long)NNODES * 64, B), B, 0, stream>>>(out2, b2, batch, pool,
                                                           NNODES * 64);
    fc_head<<<nblk(NGRAPHS, B), B, 0, stream>>>(pool, fcw, fcb, out);
}
